// Head_45561013076111
// MI455X (gfx1250) — compile-verified, matrix operand order checked by reading, not run on hardware
//
#include <hip/hip_runtime.h>

typedef __attribute__((ext_vector_type(16))) __bf16 bf16x16;
typedef __attribute__((ext_vector_type(8)))  float  f32x8;

#define B_ 4
#define T_ 2048
#define C_ 1024
#define H_ 64
#define KCHUNK 256

static __device__ __forceinline__ f32x8 wmma_bf16(bf16x16 a, bf16x16 b, f32x8 c) {
    return __builtin_amdgcn_wmma_f32_16x16x32_bf16(false, a, false, b, (short)0, c,
                                                   false, false);
}

// ---------------------------------------------------------------------------
// Kernel 1: q/k/v = x @ W{q,k,v}.  8 waves/block, 16 rows per wave (128/block).
// Weights staged in LDS as bf16 transposed [n][k] in 32KB chunks.
// ---------------------------------------------------------------------------
__global__ __launch_bounds__(256) void qkv_kernel(
    const float* __restrict__ x,
    const float* __restrict__ Wq, const float* __restrict__ Wk,
    const float* __restrict__ Wv,
    float* __restrict__ out_k, float* __restrict__ out_v,
    __bf16* __restrict__ qb, __bf16* __restrict__ kb, __bf16* __restrict__ vt)
{
    __shared__ alignas(64) __bf16 wt[H_ * KCHUNK];   // 32 KB, [n][k-kc]
    const int tid   = threadIdx.x;
    const int wave  = tid >> 5;
    const int lane  = tid & 31;
    const int l16   = lane & 15;
    const int g     = lane >> 4;         // 0/1: K-half / M-half selector
    const int khalf = g << 4;
    const int rowbase = blockIdx.x * 128 + wave * 16;

    const float* Ws[3] = {Wq, Wk, Wv};

    for (int s = 0; s < 3; ++s) {
        const float* W = Ws[s];
        f32x8 acc[4];
        #pragma unroll
        for (int nt = 0; nt < 4; ++nt)
            #pragma unroll
            for (int r = 0; r < 8; ++r) acc[nt][r] = 0.f;

        for (int kc = 0; kc < C_; kc += KCHUNK) {
            __syncthreads();
            // cooperative load: W[kc+k][n] (f32) -> wt[n][k] (bf16)
            for (int e = tid; e < KCHUNK * H_; e += 256) {
                int k = e >> 6, n = e & 63;
                wt[n * KCHUNK + k] = (__bf16)W[(size_t)(kc + k) * H_ + n];
            }
            __syncthreads();

            for (int kk = 0; kk < KCHUNK; kk += 32) {
                // A operand: 16 contiguous bf16 of row (rowbase+l16), K-half by g
                const float* ap =
                    x + (size_t)(rowbase + l16) * C_ + kc + kk + khalf;
                bf16x16 a;
                #pragma unroll
                for (int e = 0; e < 16; ++e) a[e] = (__bf16)ap[e];
                #pragma unroll
                for (int nt = 0; nt < 4; ++nt) {
                    const bf16x16 b =
                        *(const bf16x16*)&wt[(nt * 16 + l16) * KCHUNK + kk + khalf];
                    acc[nt] = wmma_bf16(a, b, acc[nt]);
                }
            }
        }
        // C/D layout: vgpr r, lane -> M = r + 8*g, N = l16
        #pragma unroll
        for (int nt = 0; nt < 4; ++nt) {
            #pragma unroll
            for (int r = 0; r < 8; ++r) {
                int m   = r + (g << 3);
                int row = rowbase + m;
                int col = nt * 16 + l16;
                float val = acc[nt][r];
                if (s == 0) {
                    qb[(size_t)row * H_ + col] = (__bf16)val;
                } else if (s == 1) {
                    out_k[(size_t)row * H_ + col] = val;
                    kb[(size_t)row * H_ + col]    = (__bf16)val;
                } else {
                    out_v[(size_t)row * H_ + col] = val;
                    int bb = row >> 11, t = row & (T_ - 1);
                    vt[((size_t)bb * H_ + col) * T_ + t] = (__bf16)val;  // [b][h][t]
                }
            }
        }
    }
}

// ---------------------------------------------------------------------------
// Kernel 2: flash attention, 1 wave per 16-row q tile, 32-key tiles, causal.
// ---------------------------------------------------------------------------
__global__ __launch_bounds__(32) void attn_kernel(
    const __bf16* __restrict__ qb, const __bf16* __restrict__ kb,
    const __bf16* __restrict__ vt, float* __restrict__ out)
{
    __shared__ alignas(64) __bf16 pl[16 * 32];   // P scratch for A re-layout
    const int lane  = threadIdx.x & 31;
    const int l16   = lane & 15;
    const int g     = lane >> 4;
    const int khalf = g << 4;
    const int qt    = blockIdx.x & (T_ / 16 - 1);
    const int b     = blockIdx.x >> 7;
    const int qbase = qt * 16;
    const size_t rowoff = (size_t)b * T_ + qbase;

    // q tile A operands for K-steps 0 and 32 (head dim 64)
    bf16x16 qa[2];
    #pragma unroll
    for (int h = 0; h < 2; ++h)
        qa[h] = *(const bf16x16*)(qb + (rowoff + l16) * H_ + h * 32 + khalf);

    float mrow[8], lrow[8];
    #pragma unroll
    for (int r = 0; r < 8; ++r) { mrow[r] = -__builtin_inff(); lrow[r] = 0.f; }
    f32x8 o[4];
    #pragma unroll
    for (int nt = 0; nt < 4; ++nt)
        #pragma unroll
        for (int r = 0; r < 8; ++r) o[nt][r] = 0.f;

    const float scale = 0.03125f;          // C**-0.5, C = 1024
    const int ntiles = (qbase + 15) / 32 + 1;

    for (int jt = 0; jt < ntiles; ++jt) {
        const int j = jt * 32;
        f32x8 s0, s1;
        #pragma unroll
        for (int r = 0; r < 8; ++r) { s0[r] = 0.f; s1[r] = 0.f; }
        // S = q @ k^T : B operand = row of kb (contiguous head dim)
        #pragma unroll
        for (int h = 0; h < 2; ++h) {
            const bf16x16 b0 = *(const bf16x16*)(kb +
                ((size_t)b * T_ + j + l16) * H_ + h * 32 + khalf);
            const bf16x16 b1 = *(const bf16x16*)(kb +
                ((size_t)b * T_ + j + 16 + l16) * H_ + h * 32 + khalf);
            s0 = wmma_bf16(qa[h], b0, s0);
            s1 = wmma_bf16(qa[h], b1, s1);
        }
        const bool need_mask = (j + 31 > qbase);
        #pragma unroll
        for (int r = 0; r < 8; ++r) {
            const int qi = qbase + r + (g << 3);
            float e0 = s0[r] * scale, e1 = s1[r] * scale;
            if (need_mask) {
                if (j + l16 > qi)      e0 = -__builtin_inff();
                if (j + 16 + l16 > qi) e1 = -__builtin_inff();
            }
            // row max over 32 keys (16 lanes of this half hold the row)
            float mx = fmaxf(e0, e1);
            #pragma unroll
            for (int d = 1; d < 16; d <<= 1) mx = fmaxf(mx, __shfl_xor(mx, d, 32));
            const float mnew  = fmaxf(mrow[r], mx);
            const float alpha = __expf(mrow[r] - mnew);
            const float p0 = __expf(e0 - mnew);
            const float p1 = __expf(e1 - mnew);
            // stash P (bf16) row-major [m][32] for A-operand reload
            pl[(r + (g << 3)) * 32 + l16]      = (__bf16)p0;
            pl[(r + (g << 3)) * 32 + 16 + l16] = (__bf16)p1;
            float rs = p0 + p1;
            #pragma unroll
            for (int d = 1; d < 16; d <<= 1) rs += __shfl_xor(rs, d, 32);
            lrow[r] = lrow[r] * alpha + rs;
            mrow[r] = mnew;
            #pragma unroll
            for (int nt = 0; nt < 4; ++nt) o[nt][r] *= alpha;
        }
        asm volatile("s_wait_dscnt 0x0" ::: "memory");
        // A operand for P (16x32): lane = row l16, 16 contiguous keys per half
        const bf16x16 pa = *(const bf16x16*)&pl[l16 * 32 + khalf];
        // B operand for V (32x16): vt[b][h][t], contiguous along t
        #pragma unroll
        for (int nt = 0; nt < 4; ++nt) {
            const bf16x16 vb = *(const bf16x16*)(vt +
                ((size_t)b * H_ + nt * 16 + l16) * T_ + j + khalf);
            o[nt] = wmma_bf16(pa, vb, o[nt]);
        }
    }
    // normalize + store f32
    #pragma unroll
    for (int nt = 0; nt < 4; ++nt) {
        #pragma unroll
        for (int r = 0; r < 8; ++r) {
            const int m = r + (g << 3);
            out[(rowoff + m) * H_ + nt * 16 + l16] = o[nt][r] / lrow[r];
        }
    }
}

extern "C" void kernel_launch(void* const* d_in, const int* in_sizes, int n_in,
                              void* d_out, int out_size, void* d_ws, size_t ws_size,
                              hipStream_t stream) {
    const float* x  = (const float*)d_in[0];
    const float* Wq = (const float*)d_in[1];
    const float* Wk = (const float*)d_in[2];
    const float* Wv = (const float*)d_in[3];

    const size_t NTH = (size_t)B_ * T_ * H_;
    float* out   = (float*)d_out;        // segment 0: out
    float* out_k = out + NTH;            // segment 1: k
    float* out_v = out_k + NTH;          // segment 2: v

    __bf16* qb = (__bf16*)d_ws;          // bf16 q, row-major [B*T][H]
    __bf16* kb = qb + NTH;               // bf16 k, row-major [B*T][H]
    __bf16* vt = kb + NTH;               // bf16 v, transposed [B][H][T]

    qkv_kernel<<<(B_ * T_) / 128, 256, 0, stream>>>(x, Wq, Wk, Wv,
                                                    out_k, out_v, qb, kb, vt);
    attn_kernel<<<B_ * (T_ / 16), 32, 0, stream>>>(qb, kb, vt, out);
}
